// PENN_24721831756522
// MI455X (gfx1250) — compile-verified, matrix operand order checked by reading, not run on hardware
//
#include <hip/hip_runtime.h>
#include <hip/hip_bf16.h>

// ---------------------------------------------------------------------------
// CDNA5 (gfx1250) GNN forward. All GEMM cores use v_wmma_f32_16x16x32_bf16
// (bf16 A/B, fp32 accumulate). Weights pre-packed into WMMA B-lane layout.
// ---------------------------------------------------------------------------

typedef __attribute__((ext_vector_type(16))) __bf16 v16bf;
typedef __attribute__((ext_vector_type(8)))  float  v8f;

#define HID 64
#define NGR 64
#define NNODES 50000
#define NEDGES 400000

// ---------------------------------------------------------------------------
// Weight packing: fp32 W[din,dout] -> bf16 B-tiles in WMMA per-lane layout.
// Tile (kt,nt): 32 lanes x 16 bf16.  lane: col = nt*16 + (lane&15),
// khalf = (lane>>4)*8 ; j<8 -> k = kt*32+khalf+j ; j>=8 -> k = kt*32+16+khalf+(j-8)
// Zero-padded outside [din,dout] so tiny-K matrices (din=4,1) just work.
// ---------------------------------------------------------------------------
__global__ void pack_w_kernel(const float* __restrict__ W, int din, int dout,
                              int NT, __bf16* __restrict__ out) {
  const int tile = blockIdx.x;          // kt*NT + nt
  const int lane = threadIdx.x;
  const int kt = tile / NT, nt = tile % NT;
  const int col = nt * 16 + (lane & 15);
  const int khalf = (lane >> 4) * 8;
  __bf16* p = out + ((size_t)tile * 32 + lane) * 16;
#pragma unroll
  for (int j = 0; j < 8; ++j) {
    int k = kt * 32 + khalf + j;
    float v = (k < din && col < dout) ? W[(size_t)k * dout + col] : 0.0f;
    p[j] = (__bf16)v;
  }
#pragma unroll
  for (int j = 0; j < 8; ++j) {
    int k = kt * 32 + 16 + khalf + j;
    float v = (k < din && col < dout) ? W[(size_t)k * dout + col] : 0.0f;
    p[8 + j] = (__bf16)v;
  }
}

// ---------------------------------------------------------------------------
// A-tile builders. ASRC: 0=contiguous[row*lda], 1=msg concat(h[src],h[dst],e),
// 2=upd concat(h,agg), 3=tiny guarded elementwise (din=4 or 1, K padded to 32).
// Chunks of 8 are segment-safe for modes 0/1/2 (boundaries 64/128 are 8-aligned).
// ---------------------------------------------------------------------------
template <int ASRC, int KT>
__device__ inline void build_a_tiles(v16bf* a, const float* __restrict__ A0,
                                     const float* __restrict__ A1,
                                     const int* __restrict__ idx0,
                                     const int* __restrict__ idx1,
                                     int din, int lda, int mtile, int lane) {
  const int row = mtile * 16 + (lane & 15);
  const int khalf = (lane >> 4) * 8;
#pragma unroll
  for (int kt = 0; kt < KT; ++kt) {
#pragma unroll
    for (int h2 = 0; h2 < 2; ++h2) {
      const int kb = kt * 32 + h2 * 16 + khalf;
      if constexpr (ASRC == 3) {
#pragma unroll
        for (int j = 0; j < 8; ++j) {
          int k = kb + j;
          a[kt][h2 * 8 + j] = (__bf16)((k < din) ? A0[(size_t)row * din + k] : 0.0f);
        }
      } else {
        const float* p;
        if constexpr (ASRC == 0) {
          p = A0 + (size_t)row * lda + kb;
        } else if constexpr (ASRC == 1) {
          if (kb < 64)       p = A0 + (size_t)idx0[row] * HID + kb;
          else if (kb < 128) p = A0 + (size_t)idx1[row] * HID + (kb - 64);
          else               p = A1 + (size_t)row * 32 + (kb - 128);
        } else { // ASRC == 2
          p = (kb < 64) ? (A0 + (size_t)row * HID + kb)
                        : (A1 + (size_t)row * HID + (kb - 64));
        }
#pragma unroll
        for (int j = 0; j < 8; ++j) a[kt][h2 * 8 + j] = (__bf16)p[j];
      }
    }
  }
}

// ---------------------------------------------------------------------------
// Pass 1 of an MLP2: h1 = A @ W1 + b1, plus per-column sum / sum-of-squares
// (BatchNorm batch stats) reduced through LDS then one global atomic per col.
// Block = 256 threads = 8 waves, each wave owns one 16-row M-tile.
// ---------------------------------------------------------------------------
template <int ASRC, int KT, int NT>
__global__ __launch_bounds__(256) void mlp_pass1(
    const float* __restrict__ A0, const float* __restrict__ A1,
    const int* __restrict__ idx0, const int* __restrict__ idx1,
    int din, int lda, int M, const __bf16* __restrict__ W1p,
    const float* __restrict__ b1, float* __restrict__ h1,
    float* __restrict__ ssum, float* __restrict__ ssq) {
  __shared__ float s_sum[NT * 16];
  __shared__ float s_sq[NT * 16];
  const int lane = threadIdx.x & 31;
  const int wv = threadIdx.x >> 5;
  if (threadIdx.x < NT * 16) { s_sum[threadIdx.x] = 0.f; s_sq[threadIdx.x] = 0.f; }
  __syncthreads();

  const int mtile = blockIdx.x * 8 + wv;
  if (mtile * 16 < M) {
    v16bf a[KT];
    build_a_tiles<ASRC, KT>(a, A0, A1, idx0, idx1, din, lda, mtile, lane);
    v8f acc[NT];
#pragma unroll
    for (int nt = 0; nt < NT; ++nt)
#pragma unroll
      for (int r = 0; r < 8; ++r) acc[nt][r] = 0.f;
#pragma unroll
    for (int nt = 0; nt < NT; ++nt) {
#pragma unroll
      for (int kt = 0; kt < KT; ++kt) {
        const v16bf b = *(const v16bf*)(W1p + (((size_t)(kt * NT + nt)) * 32 + lane) * 16);
        acc[nt] = __builtin_amdgcn_wmma_f32_16x16x32_bf16(
            false, a[kt], false, b, (short)0, acc[nt], false, false);
      }
    }
    const int d1 = NT * 16;
#pragma unroll
    for (int nt = 0; nt < NT; ++nt) {
      const int col = nt * 16 + (lane & 15);
      const float bb = b1[col];
      float s = 0.f, q = 0.f;
#pragma unroll
      for (int r = 0; r < 8; ++r) {
        const int row = mtile * 16 + (lane >> 4) * 8 + r;
        const float v = acc[nt][r] + bb;
        h1[(size_t)row * d1 + col] = v;
        s += v; q += v * v;
      }
      atomicAdd(&s_sum[col], s);
      atomicAdd(&s_sq[col], q);
    }
  }
  __syncthreads();
  if (threadIdx.x < NT * 16) {
    atomicAdd(&ssum[threadIdx.x], s_sum[threadIdx.x]);
    atomicAdd(&ssq[threadIdx.x], s_sq[threadIdx.x]);
  }
}

__global__ void bn_finalize(const float* __restrict__ ssum,
                            const float* __restrict__ ssq, int d1, float Minv,
                            float* __restrict__ mu, float* __restrict__ rstd) {
  int i = blockIdx.x * blockDim.x + threadIdx.x;
  if (i < d1) {
    float m = ssum[i] * Minv;
    mu[i] = m;
    float v = ssq[i] * Minv - m * m;
    rstd[i] = rsqrtf(v + 1e-5f);
  }
}

// ---------------------------------------------------------------------------
// Pass 2: out = relu(BN(h1)) @ W2 + b2.  SCATTER=true fuses the edge->node
// segment_sum: results atomic-added into out[sidx[row]*ldo + col].
// ---------------------------------------------------------------------------
template <int KT, int NT, bool SCATTER>
__global__ __launch_bounds__(256) void mlp_pass2(
    const float* __restrict__ h1, int M, const float* __restrict__ mu,
    const float* __restrict__ rstd, const float* __restrict__ g,
    const float* __restrict__ bt, const __bf16* __restrict__ W2p,
    const float* __restrict__ b2, float* __restrict__ out, int ldo,
    const int* __restrict__ sidx) {
  const int lane = threadIdx.x & 31;
  const int wv = threadIdx.x >> 5;
  const int mtile = blockIdx.x * 8 + wv;
  if (mtile * 16 >= M) return;
  const int d1 = KT * 32;
  const int rowA = mtile * 16 + (lane & 15);
  const int khalf = (lane >> 4) * 8;
  v16bf a[KT];
#pragma unroll
  for (int kt = 0; kt < KT; ++kt) {
#pragma unroll
    for (int h2 = 0; h2 < 2; ++h2) {
      const int kb = kt * 32 + h2 * 16 + khalf;
      const float* p = h1 + (size_t)rowA * d1 + kb;
#pragma unroll
      for (int j = 0; j < 8; ++j) {
        const int k = kb + j;
        float v = (p[j] - mu[k]) * rstd[k] * g[k] + bt[k];
        v = v > 0.f ? v : 0.f;
        a[kt][h2 * 8 + j] = (__bf16)v;
      }
    }
  }
  v8f acc[NT];
#pragma unroll
  for (int nt = 0; nt < NT; ++nt)
#pragma unroll
    for (int r = 0; r < 8; ++r) acc[nt][r] = 0.f;
#pragma unroll
  for (int nt = 0; nt < NT; ++nt) {
#pragma unroll
    for (int kt = 0; kt < KT; ++kt) {
      const v16bf b = *(const v16bf*)(W2p + (((size_t)(kt * NT + nt)) * 32 + lane) * 16);
      acc[nt] = __builtin_amdgcn_wmma_f32_16x16x32_bf16(
          false, a[kt], false, b, (short)0, acc[nt], false, false);
    }
  }
#pragma unroll
  for (int nt = 0; nt < NT; ++nt) {
    const int col = nt * 16 + (lane & 15);
    const float bb = b2[col];
#pragma unroll
    for (int r = 0; r < 8; ++r) {
      const int row = mtile * 16 + (lane >> 4) * 8 + r;
      const float v = acc[nt][r] + bb;
      if constexpr (SCATTER)
        atomicAdd(&out[(size_t)sidx[row] * ldo + col], v);
      else
        out[(size_t)row * ldo + col] = v;
    }
  }
}

// Residual + LayerNorm over HID=64 features; one 64-thread block per node row.
__global__ void residual_ln(float* __restrict__ h, const float* __restrict__ hn,
                            const float* __restrict__ g,
                            const float* __restrict__ b) {
  __shared__ float s1[64];
  __shared__ float s2[64];
  const int row = blockIdx.x, t = threadIdx.x;
  const float v = h[(size_t)row * HID + t] + hn[(size_t)row * HID + t];
  s1[t] = v; s2[t] = v * v;
  __syncthreads();
  for (int off = 32; off > 0; off >>= 1) {
    if (t < off) { s1[t] += s1[t + off]; s2[t] += s2[t + off]; }
    __syncthreads();
  }
  const float mu = s1[0] * (1.0f / 64.0f);
  const float var = s2[0] * (1.0f / 64.0f) - mu * mu;
  const float rs = rsqrtf(var + 1e-5f);
  h[(size_t)row * HID + t] = (v - mu) * rs * g[t] + b[t];
}

// Graph add-pool: atomic scatter into hg[batch[n], :].
__global__ void pool_kernel(const float* __restrict__ h,
                            const int* __restrict__ batch,
                            float* __restrict__ hg, int N) {
  int i = blockIdx.x * blockDim.x + threadIdx.x;
  if (i < N * HID) {
    int row = i >> 6, c = i & 63;
    atomicAdd(&hg[(size_t)batch[row] * HID + c], h[i]);
  }
}

// Head pass 2: relu(BN(h1[64,32])) @ W2[32,1] + b2 -> out[64].
__global__ void head_pass2(const float* __restrict__ h1,
                           const float* __restrict__ mu,
                           const float* __restrict__ rstd,
                           const float* __restrict__ g,
                           const float* __restrict__ bt,
                           const float* __restrict__ W2,
                           const float* __restrict__ b2,
                           float* __restrict__ out) {
  int row = threadIdx.x;
  if (row < NGR) {
    float acc = b2[0];
    for (int k = 0; k < 32; ++k) {
      float v = (h1[row * 32 + k] - mu[k]) * rstd[k] * g[k] + bt[k];
      v = v > 0.f ? v : 0.f;
      acc += v * W2[k];
    }
    out[row] = acc;
  }
}

// ---------------------------------------------------------------------------
// Host side
// ---------------------------------------------------------------------------
struct MlpP { const float *W1, *W2, *b1, *b2, *bt, *g; };

extern "C" void kernel_launch(void* const* d_in, const int* in_sizes, int n_in,
                              void* d_out, int out_size, void* d_ws,
                              size_t ws_size, hipStream_t stream) {
  (void)in_sizes; (void)n_in; (void)out_size; (void)ws_size;
  const int N = NNODES, E = NEDGES;

  auto f = [&](int i) { return (const float*)d_in[i]; };
  const float* x         = f(0);
  const int*   src       = (const int*)d_in[1];
  const int*   dst       = src + E;
  const float* edge_attr = f(2);
  const int*   batch     = (const int*)d_in[3];
  // JAX pytree leaf order: dict keys sorted. params keys:
  // edge_enc(4..9), head(10..15), layers[0..3] (14 leaves each), node_enc(72..77)
  // mlp2 leaf order: W1, W2, b1, b2, bt, g ; layer: ln_b, ln_g, msg(6), upd(6)
  auto mlp_at = [&](int i) { return MlpP{f(i), f(i + 1), f(i + 2), f(i + 3), f(i + 4), f(i + 5)}; };
  MlpP edge_enc = mlp_at(4);
  MlpP head     = mlp_at(10);
  const float* ln_b[4]; const float* ln_g[4]; MlpP msg[4]; MlpP upd[4];
  for (int l = 0; l < 4; ++l) {
    int base = 16 + 14 * l;
    ln_b[l] = f(base); ln_g[l] = f(base + 1);
    msg[l] = mlp_at(base + 2);
    upd[l] = mlp_at(base + 8);
  }
  MlpP node_enc = mlp_at(72);

  // ---- workspace layout (floats) ----
  size_t o = 0;
  auto alloc = [&](size_t n) { float* p = (float*)d_ws + o; o += n; return p; };
  const size_t N64 = (size_t)N * HID;
  float* h    = alloc(N64);
  float* hnew = alloc(N64);
  float* agg  = alloc(N64);
  float* h1n  = alloc(N64);
  float* eenc = alloc((size_t)E * 32);
  float* h1e  = alloc((size_t)E * 64);
  float* hg   = alloc(64 * 64);
  float* h1h  = alloc(64 * 32);
  float* ssum = alloc(64);
  float* ssq  = alloc(64);   // contiguous with ssum: one memset clears both
  float* mu   = alloc(64);
  float* rstd = alloc(64);
  o = (o + 63) & ~(size_t)63;  // 256B align for v16bf loads
  __bf16* pw = (__bf16*)((float*)d_ws + o);

  size_t po = 0;
  auto palloc = [&](int KT, int NT) { __bf16* p = pw + po; po += (size_t)KT * NT * 512; return p; };
  __bf16* p_nW1 = palloc(1, 4);  __bf16* p_nW2 = palloc(2, 4);
  __bf16* p_eW1 = palloc(1, 2);  __bf16* p_eW2 = palloc(1, 2);
  __bf16* p_hW1 = palloc(2, 2);
  __bf16 *p_mW1[4], *p_mW2[4], *p_uW1[4], *p_uW2[4];
  for (int l = 0; l < 4; ++l) {
    p_mW1[l] = palloc(5, 4); p_mW2[l] = palloc(2, 4);
    p_uW1[l] = palloc(4, 4); p_uW2[l] = palloc(2, 4);
  }

  // ---- pack all weights into WMMA B layout ----
  pack_w_kernel<<<1 * 4, 32, 0, stream>>>(node_enc.W1, 4, 64, 4, p_nW1);
  pack_w_kernel<<<2 * 4, 32, 0, stream>>>(node_enc.W2, 64, 64, 4, p_nW2);
  pack_w_kernel<<<1 * 2, 32, 0, stream>>>(edge_enc.W1, 1, 32, 2, p_eW1);
  pack_w_kernel<<<1 * 2, 32, 0, stream>>>(edge_enc.W2, 32, 32, 2, p_eW2);
  pack_w_kernel<<<2 * 2, 32, 0, stream>>>(head.W1, 64, 32, 2, p_hW1);
  for (int l = 0; l < 4; ++l) {
    pack_w_kernel<<<5 * 4, 32, 0, stream>>>(msg[l].W1, 160, 64, 4, p_mW1[l]);
    pack_w_kernel<<<2 * 4, 32, 0, stream>>>(msg[l].W2, 64, 64, 4, p_mW2[l]);
    pack_w_kernel<<<4 * 4, 32, 0, stream>>>(upd[l].W1, 128, 64, 4, p_uW1[l]);
    pack_w_kernel<<<2 * 4, 32, 0, stream>>>(upd[l].W2, 64, 64, 4, p_uW2[l]);
  }

  const int gN = (N + 127) / 128;   // 8 waves x 16 rows per block
  const int gE = (E + 127) / 128;

  // ---- node encoder ----
  hipMemsetAsync(ssum, 0, 128 * sizeof(float), stream);
  mlp_pass1<3, 1, 4><<<gN, 256, 0, stream>>>(x, nullptr, nullptr, nullptr, 4, 4,
                                             N, p_nW1, node_enc.b1, h1n, ssum, ssq);
  bn_finalize<<<1, 64, 0, stream>>>(ssum, ssq, 64, 1.0f / N, mu, rstd);
  mlp_pass2<2, 4, false><<<gN, 256, 0, stream>>>(h1n, N, mu, rstd, node_enc.g,
                                                 node_enc.bt, p_nW2, node_enc.b2,
                                                 h, HID, nullptr);
  // ---- edge encoder ----
  hipMemsetAsync(ssum, 0, 128 * sizeof(float), stream);
  mlp_pass1<3, 1, 2><<<gE, 256, 0, stream>>>(edge_attr, nullptr, nullptr, nullptr,
                                             1, 1, E, p_eW1, edge_enc.b1, h1e,
                                             ssum, ssq);
  bn_finalize<<<1, 64, 0, stream>>>(ssum, ssq, 32, 1.0f / E, mu, rstd);
  mlp_pass2<1, 2, false><<<gE, 256, 0, stream>>>(h1e, E, mu, rstd, edge_enc.g,
                                                 edge_enc.bt, p_eW2, edge_enc.b2,
                                                 eenc, 32, nullptr);
  // ---- message-passing layers ----
  for (int l = 0; l < 4; ++l) {
    hipMemsetAsync(agg, 0, N64 * sizeof(float), stream);
    hipMemsetAsync(ssum, 0, 128 * sizeof(float), stream);
    mlp_pass1<1, 5, 4><<<gE, 256, 0, stream>>>(h, eenc, src, dst, 160, 0, E,
                                               p_mW1[l], msg[l].b1, h1e, ssum, ssq);
    bn_finalize<<<1, 64, 0, stream>>>(ssum, ssq, 64, 1.0f / E, mu, rstd);
    mlp_pass2<2, 4, true><<<gE, 256, 0, stream>>>(h1e, E, mu, rstd, msg[l].g,
                                                  msg[l].bt, p_mW2[l], msg[l].b2,
                                                  agg, HID, dst);
    hipMemsetAsync(ssum, 0, 128 * sizeof(float), stream);
    mlp_pass1<2, 4, 4><<<gN, 256, 0, stream>>>(h, agg, nullptr, nullptr, 128, 0,
                                               N, p_uW1[l], upd[l].b1, h1n, ssum, ssq);
    bn_finalize<<<1, 64, 0, stream>>>(ssum, ssq, 64, 1.0f / N, mu, rstd);
    mlp_pass2<2, 4, false><<<gN, 256, 0, stream>>>(h1n, N, mu, rstd, upd[l].g,
                                                   upd[l].bt, p_uW2[l], upd[l].b2,
                                                   hnew, HID, nullptr);
    residual_ln<<<N, 64, 0, stream>>>(h, hnew, ln_g[l], ln_b[l]);
  }
  // ---- pooling + head ----
  hipMemsetAsync(hg, 0, 64 * 64 * sizeof(float), stream);
  pool_kernel<<<(N * HID + 255) / 256, 256, 0, stream>>>(h, batch, hg, N);
  hipMemsetAsync(ssum, 0, 128 * sizeof(float), stream);
  mlp_pass1<0, 2, 2><<<1, 256, 0, stream>>>(hg, nullptr, nullptr, nullptr, 64, 64,
                                            64, p_hW1, head.b1, h1h, ssum, ssq);
  bn_finalize<<<1, 64, 0, stream>>>(ssum, ssq, 32, 1.0f / 64.0f, mu, rstd);
  head_pass2<<<1, 64, 0, stream>>>(h1h, mu, rstd, head.g, head.bt, head.W2,
                                   head.b2, (float*)d_out);
}
